// FeedforwardSNN_48301202211068
// MI455X (gfx1250) — compile-verified, matrix operand order checked by reading, not run on hardware
//
#include <hip/hip_runtime.h>
#include <hip/hip_bf16.h>

// ---------------------------------------------------------------------------
// FeedforwardSNN for MI455X (gfx1250, wave32, WMMA)
//
// Phase 1: convert spikes f32->f16 (exact: values are 0/1) and split weight
//          into W_hi(f16) + W_lo(f16 residual)  (Dekker split ~ fp32 accuracy)
// Phase 2: register-blocked WMMA GEMM  cur[M=16384, N=4096], K=2048
//          using V_WMMA_F32_16X16X32_F16, fragments loaded directly from
//          global (entire f16 working set ~96MB fits the 192MB L2)
// Phase 3: sequential LIF scan over T: bandwidth-bound (768MB -> ~33us floor
//          at 23.3 TB/s), 4-wide b128 accesses with non-temporal hints
// ---------------------------------------------------------------------------

typedef __attribute__((ext_vector_type(16))) _Float16 v16h;
typedef __attribute__((ext_vector_type(8)))  _Float16 v8h;
typedef __attribute__((ext_vector_type(4)))  _Float16 v4h;
typedef __attribute__((ext_vector_type(8)))  float    v8f;
typedef __attribute__((ext_vector_type(4)))  float    v4f;

constexpr int T_STEPS = 256;
constexpr int BATCH   = 64;
constexpr int N_IN    = 2048;   // K
constexpr int N_HID   = 4096;   // N
constexpr int M_ROWS  = T_STEPS * BATCH;  // 16384

constexpr float BETA         = 0.9f;
constexpr float THRESH       = 1.0f;
constexpr float REFRAC_STEPS = 3.0f;

// workspace layout (bytes)
constexpr size_t CUR_BYTES    = (size_t)M_ROWS * N_HID * sizeof(float);     // 256 MB
constexpr size_t SPIKE_BYTES  = (size_t)M_ROWS * N_IN  * sizeof(_Float16);  //  64 MB
constexpr size_t WSPLIT_BYTES = (size_t)N_HID  * N_IN  * sizeof(_Float16);  //  16 MB each

union FragAB { v16h v; v8h h[2]; };

// --------------------------- conversion kernels ----------------------------
// One-shot, bandwidth-bound: 4-wide, NT loads (inputs re-read later only in
// f16 form) and NT stores would hurt (f16 arrays ARE re-read by the GEMM),
// so stores stay regular-temporal to seed L2.

__global__ void cvt_f32_to_f16(const float* __restrict__ in,
                               _Float16* __restrict__ out, int n4) {
  int i = blockIdx.x * blockDim.x + threadIdx.x;
  int stride = gridDim.x * blockDim.x;
  for (; i < n4; i += stride) {
    v4f x = __builtin_nontemporal_load((const v4f*)in + i);
    v4h h;
    h.x = (_Float16)x.x; h.y = (_Float16)x.y;
    h.z = (_Float16)x.z; h.w = (_Float16)x.w;
    ((v4h*)out)[i] = h;
  }
}

__global__ void split_weight_f16(const float* __restrict__ w,
                                 _Float16* __restrict__ hi,
                                 _Float16* __restrict__ lo, int n4) {
  int i = blockIdx.x * blockDim.x + threadIdx.x;
  int stride = gridDim.x * blockDim.x;
  for (; i < n4; i += stride) {
    v4f x = __builtin_nontemporal_load((const v4f*)w + i);
    v4h h, l;
    h.x = (_Float16)x.x; l.x = (_Float16)(x.x - (float)h.x);
    h.y = (_Float16)x.y; l.y = (_Float16)(x.y - (float)h.y);
    h.z = (_Float16)x.z; l.z = (_Float16)(x.z - (float)h.z);
    h.w = (_Float16)x.w; l.w = (_Float16)(x.w - (float)h.w);
    ((v4h*)hi)[i] = h;
    ((v4h*)lo)[i] = l;
  }
}

// ------------------------------- WMMA GEMM ---------------------------------
// C[M,N] = A[M,K] * W^T where W is [N,K] row-major (so B columns are W rows,
// contiguous along K -> same fragment load pattern as A; no transpose needed).
//
// Block: 256 threads = 8 waves arranged 4(M) x 2(N).
// Wave tile: 64x32 = 4x2 fragments of 16x16. Block tile: 256x64.
// K loop: 2048 / 32 = 64 iterations, 16 v_wmma per iteration per wave
// (hi+lo split -> near-fp32 product accuracy into f32 accumulators).

__global__ __launch_bounds__(256)
void snn_gemm_wmma(const _Float16* __restrict__ A,
                   const _Float16* __restrict__ Whi,
                   const _Float16* __restrict__ Wlo,
                   float* __restrict__ C) {
  const int lane  = threadIdx.x & 31;
  const int wave  = threadIdx.x >> 5;
  const int waveM = wave & 3;   // 0..3
  const int waveN = wave >> 2;  // 0..1
  const int mWave = blockIdx.x * 256 + waveM * 64;
  const int nWave = blockIdx.y * 64  + waveN * 32;

  // A/B 16-bit fragment addressing (ISA 7.12.2):
  //   lane L: row/col = L%16 ; K chunk select = (L/16)*8
  //   VGPR0..3 <- K = khalf..khalf+7 ; VGPR4..7 <- K = 16+khalf..16+khalf+7
  const int r16   = lane & 15;
  const int khalf = (lane >> 4) << 3;

  const _Float16* aPtr[4];
#pragma unroll
  for (int i = 0; i < 4; ++i)
    aPtr[i] = A + (size_t)(mWave + i * 16 + r16) * N_IN + khalf;

  const _Float16* bhPtr[2];
  const _Float16* blPtr[2];
#pragma unroll
  for (int j = 0; j < 2; ++j) {
    size_t off = (size_t)(nWave + j * 16 + r16) * N_IN + khalf;
    bhPtr[j] = Whi + off;
    blPtr[j] = Wlo + off;
  }

  v8f acc[4][2] = {};

  for (int k0 = 0; k0 < N_IN; k0 += 32) {
    FragAB a[4], bh[2], bl[2];
#pragma unroll
    for (int i = 0; i < 4; ++i) {
      a[i].h[0] = *(const v8h*)(aPtr[i] + k0);
      a[i].h[1] = *(const v8h*)(aPtr[i] + k0 + 16);
    }
#pragma unroll
    for (int j = 0; j < 2; ++j) {
      bh[j].h[0] = *(const v8h*)(bhPtr[j] + k0);
      bh[j].h[1] = *(const v8h*)(bhPtr[j] + k0 + 16);
      bl[j].h[0] = *(const v8h*)(blPtr[j] + k0);
      bl[j].h[1] = *(const v8h*)(blPtr[j] + k0 + 16);
    }

    // warm next K tiles into WGP-level cache (locality 3 -> WGP scope)
    if (k0 + 96 < N_IN) {
      __builtin_prefetch(aPtr[0] + k0 + 96, 0, 3);
      __builtin_prefetch(bhPtr[0] + k0 + 96, 0, 3);
      __builtin_prefetch(blPtr[0] + k0 + 96, 0, 3);
    }

#pragma unroll
    for (int i = 0; i < 4; ++i) {
#pragma unroll
      for (int j = 0; j < 2; ++j) {
        acc[i][j] = __builtin_amdgcn_wmma_f32_16x16x32_f16(
            false, a[i].v, false, bh[j].v, (short)0, acc[i][j], false, false);
        acc[i][j] = __builtin_amdgcn_wmma_f32_16x16x32_f16(
            false, a[i].v, false, bl[j].v, (short)0, acc[i][j], false, false);
      }
    }
  }

  // D layout: lane L -> N = L%16 ; VGPR r -> M = (L/16)*8 + r
  // Regular-temporal stores: cur is re-read immediately by the LIF scan,
  // so we want it resident in L2.
  const int mOff = (lane >> 4) * 8;
  const int col  = lane & 15;
#pragma unroll
  for (int i = 0; i < 4; ++i) {
#pragma unroll
    for (int j = 0; j < 2; ++j) {
      float* cp = C + (size_t)(mWave + i * 16 + mOff) * N_HID
                    + (nWave + j * 16 + col);
#pragma unroll
      for (int r = 0; r < 8; ++r)
        cp[(size_t)r * N_HID] = acc[i][j][r];
    }
  }
}

// ------------------------------- LIF scan ----------------------------------
// One thread per 4 consecutive (b, h) elements; consecutive threads cover
// consecutive 16B chunks -> fully coalesced b128 traffic.
// cur reads are last-use -> NT loads; outputs are never re-read -> NT stores
// (keeps the 512MB output stream from thrashing L2).

__global__ __launch_bounds__(256)
void lif_scan(const float* __restrict__ cur,
              float* __restrict__ spk_out,
              float* __restrict__ mem_out) {
  const int idx4 = blockIdx.x * blockDim.x + threadIdx.x;  // 0 .. B*N_HID/4-1
  const size_t bn4 = (size_t)BATCH * N_HID / 4;

  float mem[4]    = {0.0f, 0.0f, 0.0f, 0.0f};
  float refrac[4] = {0.0f, 0.0f, 0.0f, 0.0f};

  for (int t = 0; t < T_STEPS; ++t) {
    const size_t off = (size_t)t * bn4 + idx4;
    v4f c = __builtin_nontemporal_load((const v4f*)cur + off);
    v4f s, m;
#pragma unroll
    for (int e = 0; e < 4; ++e) {
      float ce = (e == 0) ? c.x : (e == 1) ? c.y : (e == 2) ? c.z : c.w;
      float reset = (mem[e] > THRESH) ? 1.0f : 0.0f;
      mem[e] = BETA * mem[e] + ce - reset * THRESH;
      float spk_raw = (mem[e] > THRESH) ? 1.0f : 0.0f;
      float in_ref  = (refrac[e] > 0.0f) ? 1.0f : 0.0f;
      float spk     = spk_raw * (1.0f - in_ref);
      refrac[e] = fmaxf(refrac[e] - 1.0f, 0.0f);
      refrac[e] = (spk > 0.0f) ? REFRAC_STEPS : refrac[e];
      if (e == 0) { s.x = spk; m.x = mem[e]; }
      else if (e == 1) { s.y = spk; m.y = mem[e]; }
      else if (e == 2) { s.z = spk; m.z = mem[e]; }
      else { s.w = spk; m.w = mem[e]; }
    }
    __builtin_nontemporal_store(s, (v4f*)spk_out + off);
    __builtin_nontemporal_store(m, (v4f*)mem_out + off);
  }
}

// ------------------------------- launcher ----------------------------------

extern "C" void kernel_launch(void* const* d_in, const int* in_sizes, int n_in,
                              void* d_out, int out_size, void* d_ws, size_t ws_size,
                              hipStream_t stream) {
  const float* spikes = (const float*)d_in[0];  // [T, B, N_IN]  f32 (0/1)
  const float* weight = (const float*)d_in[1];  // [N_HID, N_IN] f32

  float* out_spk = (float*)d_out;                               // [T,B,N_HID]
  float* out_mem = out_spk + (size_t)T_STEPS * BATCH * N_HID;   // [T,B,N_HID]

  char* ws = (char*)d_ws;
  float*    cur  = (float*)ws;                                        // 256 MB
  _Float16* a_h  = (_Float16*)(ws + CUR_BYTES);                       //  64 MB
  _Float16* w_hi = (_Float16*)(ws + CUR_BYTES + SPIKE_BYTES);         //  16 MB
  _Float16* w_lo = (_Float16*)(ws + CUR_BYTES + SPIKE_BYTES + WSPLIT_BYTES);

  // Phase 1: precision conversion / weight split (4-wide)
  cvt_f32_to_f16<<<2048, 256, 0, stream>>>(spikes, a_h, (M_ROWS * N_IN) / 4);
  split_weight_f16<<<1024, 256, 0, stream>>>(weight, w_hi, w_lo, (N_HID * N_IN) / 4);

  // Phase 2: cur = spikes @ W^T  (M=16384, N=4096, K=2048)
  dim3 grid(M_ROWS / 256, N_HID / 64);
  snn_gemm_wmma<<<grid, 256, 0, stream>>>(a_h, w_hi, w_lo, cur);

  // Phase 3: sequential LIF dynamics over T (4-wide, NT streams)
  lif_scan<<<(BATCH * N_HID) / (4 * 256), 256, 0, stream>>>(cur, out_spk, out_mem);
}
